// SparseConvEncoder_52175262712455
// MI455X (gfx1250) — compile-verified, matrix operand order checked by reading, not run on hardware
//
#include <hip/hip_runtime.h>

typedef unsigned short u16;
typedef __attribute__((ext_vector_type(16))) __bf16 v16bf;
typedef __attribute__((ext_vector_type(8)))  float  v8f;

__device__ __forceinline__ u16 f2bf(float f) {
    unsigned u = __float_as_uint(f);
    unsigned r = u + 0x7FFFu + ((u >> 16) & 1u);   // round-to-nearest-even
    return (u16)(r >> 16);
}
__device__ __forceinline__ float bf2f(u16 h) {
    return __uint_as_float(((unsigned)h) << 16);
}

union Frag { uint4 q[2]; v16bf v; };

// ---------------------------------------------------------------------------
// Implicit-GEMM conv3d, bf16 in / WMMA f32 accumulate. All index math 32-bit.
//   in : NDHWC bf16 activations  [2, DI, HI, WI, CIN]
//   wt : transposed bf16 weights [COUT][KS^3 * CIN]
// One block = 64 consecutive output voxels x all COUT channels.
// 8 wave32 / block; wave pair shares an M subtile, each wave owns COUT/32
// 16x16 accumulator tiles and reuses one A fragment across them.
// ---------------------------------------------------------------------------
template<int CIN, int COUT, int KS, int STRIDE, int PAD,
         int DO, int HO, int WO, bool POST, bool RESID, bool OUTF32>
__global__ __launch_bounds__(256) void conv_wmma(
    const u16* __restrict__ in, const u16* __restrict__ wt,
    const float* __restrict__ ps, const float* __restrict__ pb,
    const u16* __restrict__ res, void* __restrict__ outp)
{
    constexpr int NB   = 2;
    constexpr int DI   = (KS == 3) ? DO : DO * STRIDE;
    constexpr int HI   = (KS == 3) ? HO : HO * STRIDE;
    constexpr int WI   = (KS == 3) ? WO : WO * STRIDE;
    constexpr int KS3  = KS * KS * KS;
    constexpr int KSTR = KS3 * CIN;            // reduction length (row of Wt)
    constexpr int NTPW = COUT / 32;            // N tiles per wave
    constexpr int NV   = NB * DO * HO * WO;    // total output voxels (fits i32)

    const int tid  = threadIdx.x;
    const int wave = tid >> 5, lane = tid & 31;
    const int hi   = lane >> 4, lo = lane & 15;
    const int mv   = wave >> 1;                // M subtile 0..3
    const int n0   = wave & 1;                 // N tile parity
    const int khalf = hi * 8;                  // A/B 16-bit fragment K offset

    const int vbase = blockIdx.x * 64;

    // A-row voxel for this lane (A layout: lane holds row m = lane&15)
    const int va  = vbase + mv * 16 + lo;
    const bool mval = va < NV;
    int t_ = va;
    const int ax = t_ % WO; t_ /= WO;
    const int ay = t_ % HO; t_ /= HO;
    const int az = t_ % DO;
    const int ab = t_ / DO;

    // per-lane base pointer at tap (0,0,0); may be "negative" for PAD, only
    // dereferenced when the tap is valid.
    const int bx = ax * STRIDE - PAD;
    const int by = ay * STRIDE - PAD;
    const int bz = az * STRIDE - PAD;
    const int baseoff = (((ab * DI + bz) * HI + by) * WI + bx) * CIN + khalf;
    const u16* abase = in + baseoff;

    v8f acc[NTPW] = {};

    const u16* wrow[NTPW];
#pragma unroll
    for (int t = 0; t < NTPW; ++t) {
        const int ng = (n0 + 2 * t) * 16 + lo;        // global out channel
        wrow[t] = wt + ng * KSTR + khalf;
    }

    for (int tap = 0; tap < KS3; ++tap) {
        const int dx = tap % KS, dy = (tap / KS) % KS, dz = tap / (KS * KS);
        bool valid = mval;
        if (PAD) {
            const int ix = bx + dx, iy = by + dy, iz = bz + dz;
            valid = valid && ((unsigned)ix < (unsigned)WI)
                          && ((unsigned)iy < (unsigned)HI)
                          && ((unsigned)iz < (unsigned)DI);
        }
        const u16* arow = abase + ((dz * HI + dy) * WI + dx) * CIN;

        // speculative prefetch of next tap's A row (global_prefetch_b8)
        if (tap + 1 < KS3) {
            const int tn = tap + 1;
            const int nx = tn % KS, ny = (tn / KS) % KS, nz = tn / (KS * KS);
            __builtin_prefetch(abase + ((nz * HI + ny) * WI + nx) * CIN, 0, 3);
        }

#pragma unroll
        for (int kc = 0; kc < CIN / 32; ++kc) {
            Frag a;
            if (valid) {
                a.q[0] = *(const uint4*)(arow + kc * 32);
                a.q[1] = *(const uint4*)(arow + kc * 32 + 16);
            } else {
                a.q[0] = make_uint4(0, 0, 0, 0);
                a.q[1] = make_uint4(0, 0, 0, 0);
            }
#pragma unroll
            for (int t = 0; t < NTPW; ++t) {
                Frag bfr;
                const u16* wr = wrow[t] + tap * CIN + kc * 32;
                bfr.q[0] = *(const uint4*)(wr);
                bfr.q[1] = *(const uint4*)(wr + 16);
                acc[t] = __builtin_amdgcn_wmma_f32_16x16x32_bf16(
                    false, a.v, false, bfr.v, (short)0, acc[t], false, false);
            }
        }
    }

    // Epilogue. D layout: lanes 0-15 VGPR v -> M=v; lanes 16-31 -> M=v+8; N=lane&15
#pragma unroll
    for (int t = 0; t < NTPW; ++t) {
        const int n = (n0 + 2 * t) * 16 + lo;
        float sc = 1.f, bi = 0.f;
        if (POST) { sc = ps[n]; bi = pb[n]; }
#pragma unroll
        for (int vi = 0; vi < 8; ++vi) {
            const int v = vbase + mv * 16 + hi * 8 + vi;
            if (v < NV) {
                float o = acc[t][vi];
                if (POST)  o = fmaxf(fmaf(o, sc, bi), 0.f);
                if (RESID) o += bf2f(res[v * COUT + n]);
                if (OUTF32) ((float*)outp)[v * COUT + n] = o;
                else        ((u16*)outp)[v * COUT + n] = f2bf(o);
            }
        }
    }
}

// Elementwise pre-activation, 8 bf16 per thread (uint4 in/out):
//   y = bf16(relu(x*s[c] + b[c])), C power of two >= 64.
__global__ __launch_bounds__(256) void preact_kernel(
    const u16* __restrict__ x, const float* __restrict__ s,
    const float* __restrict__ b, u16* __restrict__ y, int n8, int cmask)
{
    int i = blockIdx.x * 256 + threadIdx.x;
    if (i >= n8) return;
    union { uint4 q; u16 h[8]; } iv, ov;
    iv.q = ((const uint4*)x)[i];
    const int cb = (i * 8) & cmask;
#pragma unroll
    for (int j = 0; j < 8; ++j)
        ov.h[j] = f2bf(fmaxf(fmaf(bf2f(iv.h[j]), s[cb + j], b[cb + j]), 0.f));
    ((uint4*)y)[i] = ov.q;
}

// Transpose+convert weights: W[kk][cout] f32 -> Wt[cout][kk] bf16.
__global__ __launch_bounds__(256) void wtrans_kernel(
    const float* __restrict__ w, u16* __restrict__ wt, int K, int COUT)
{
    int i = blockIdx.x * 256 + threadIdx.x;
    if (i >= K * COUT) return;
    int cout = i / K, kk = i - cout * K;
    wt[i] = f2bf(w[kk * COUT + cout]);
}

// Stem: 3x3x3 conv, 3->32 channels, SAME, fused BN+ReLU, fp32 in -> bf16 out.
// One thread = one voxel x 8 output channels; weights staged in LDS.
__global__ __launch_bounds__(256) void stem_kernel(
    const float* __restrict__ x, const float* __restrict__ w,
    const float* __restrict__ s, const float* __restrict__ b,
    u16* __restrict__ out)
{
    __shared__ float wl[27 * 3 * 32];
    for (int i = threadIdx.x; i < 2592; i += 256) wl[i] = w[i];
    __syncthreads();

    const int gid = blockIdx.x * 256 + threadIdx.x;   // < 7,077,888
    const int g = gid & 3;                            // channel group (8 ch)
    int v = gid >> 2;
    const int X = v % 96; v /= 96;
    const int Y = v % 96; v /= 96;
    const int Z = v % 96;
    const int Bb = v / 96;

    float acc[8];
#pragma unroll
    for (int k = 0; k < 8; ++k) acc[k] = 0.f;

    for (int tap = 0; tap < 27; ++tap) {
        const int dx = tap % 3, dy = (tap / 3) % 3, dz = tap / 9;
        const int ix = X + dx - 1, iy = Y + dy - 1, iz = Z + dz - 1;
        if ((unsigned)ix >= 96u || (unsigned)iy >= 96u || (unsigned)iz >= 96u) continue;
        const float* px = x + ((((Bb * 96 + iz) * 96 + iy) * 96) + ix) * 3;
        const float x0 = px[0], x1 = px[1], x2 = px[2];
        const float* w0 = wl + tap * 96 + g * 8;
#pragma unroll
        for (int k = 0; k < 8; ++k) {
            float a = fmaf(x0, w0[k], acc[k]);
            a = fmaf(x1, w0[32 + k], a);
            acc[k] = fmaf(x2, w0[64 + k], a);
        }
    }
    union { u16 h[8]; uint4 q; } r;
    const int co = g * 8;
#pragma unroll
    for (int k = 0; k < 8; ++k)
        r.h[k] = f2bf(fmaxf(fmaf(acc[k], s[co + k], b[co + k]), 0.f));
    *(uint4*)(out + (size_t)(gid >> 2) * 32 + co) = r.q;
}

extern "C" void kernel_launch(void* const* d_in, const int* in_sizes, int n_in,
                              void* d_out, int out_size, void* d_ws, size_t ws_size,
                              hipStream_t stream)
{
    (void)in_sizes; (void)n_in; (void)out_size; (void)ws_size;
    char* ws = (char*)d_ws;
    const long MB = 1 << 20;

    // ---- bf16 transposed weights for the 12 WMMA layers (region: 0..8 MiB) --
    const int widx[12] = {4, 7, 10, 13, 16, 19, 22, 25, 28, 31, 34, 37};
    const int wK[12]   = {8*32, 27*64, 27*64, 8*64, 27*128, 27*128,
                          8*128, 27*128, 27*128, 8*128, 27*128, 27*128};
    const int wCo[12]  = {64, 64, 64, 128, 128, 128, 128, 128, 128, 128, 128, 128};
    long woff[12]; long off = 0;
    for (int i = 0; i < 12; ++i) { woff[i] = off; off += (long)wK[i] * wCo[i] * 2; }
    for (int i = 0; i < 12; ++i) {
        int n = wK[i] * wCo[i];
        wtrans_kernel<<<(n + 255) / 256, 256, 0, stream>>>(
            (const float*)d_in[widx[i]], (u16*)(ws + woff[i]), wK[i], wCo[i]);
    }

    // ---- activation buffers (bytes from ws base; reuse schedule keeps peak
    //      at 120 MiB + 27 MiB) ----------------------------------------------
    u16* A   = (u16*)(ws +   8 * MB);   // stem out  [2,96,96,96,32]  108 MiB
    u16* Bs1 = (u16*)(ws + 120 * MB);   // s1 out / res1              27 MiB
    u16* T1  = (u16*)(ws +   8 * MB);   // preact(s1)   (A dead)
    u16* R1A = (u16*)(ws +  40 * MB);   // r1a out
    u16* PRE = (u16*)(ws +  72 * MB);   // shared preact scratch (<=27 MiB)
    u16* Dq  = (u16*)(ws +   8 * MB);   // r1b out      (T1 dead)
    u16* E   = (u16*)(ws +  40 * MB);   // s2 out / res2 (R1A dead)   6.75 MiB
    u16* F   = (u16*)(ws +  50 * MB);   // r2a out
    u16* G   = (u16*)(ws +  60 * MB);   // r2b out
    u16* H   = (u16*)(ws +   8 * MB);   // s3 out / res3 (Dq dead)    0.85 MiB
    u16* I   = (u16*)(ws +  10 * MB);   // r3a out
    u16* J   = (u16*)(ws +  12 * MB);   // r3b out
    u16* K   = (u16*)(ws +  14 * MB);   // s4 out / res4              0.11 MiB
    u16* Lq  = (u16*)(ws +  16 * MB);   // r4a out

    const float* x = (const float*)d_in[0];

    // stem: 2*96^3 voxels * 4 channel-groups = 7,077,888 threads
    stem_kernel<<<27648, 256, 0, stream>>>(
        x, (const float*)d_in[1], (const float*)d_in[2], (const float*)d_in[3], A);

    // stage 1 ---------------------------------------------------------------
    conv_wmma<32, 64, 2, 2, 0, 48, 48, 48, true, false, false><<<3456, 256, 0, stream>>>(
        A, (const u16*)(ws + woff[0]),
        (const float*)d_in[5], (const float*)d_in[6], nullptr, Bs1);
    const int n1 = 2 * 48 * 48 * 48 * 64;
    preact_kernel<<<(n1 / 8 + 255) / 256, 256, 0, stream>>>(
        Bs1, (const float*)d_in[8], (const float*)d_in[9], T1, n1 / 8, 63);
    conv_wmma<64, 64, 3, 1, 1, 48, 48, 48, false, false, false><<<3456, 256, 0, stream>>>(
        T1, (const u16*)(ws + woff[1]), nullptr, nullptr, nullptr, R1A);
    preact_kernel<<<(n1 / 8 + 255) / 256, 256, 0, stream>>>(
        R1A, (const float*)d_in[11], (const float*)d_in[12], PRE, n1 / 8, 63);
    conv_wmma<64, 64, 3, 1, 1, 48, 48, 48, false, true, false><<<3456, 256, 0, stream>>>(
        PRE, (const u16*)(ws + woff[2]), nullptr, nullptr, Bs1, Dq);

    // stage 2 ---------------------------------------------------------------
    conv_wmma<64, 128, 2, 2, 0, 24, 24, 24, true, false, false><<<432, 256, 0, stream>>>(
        Dq, (const u16*)(ws + woff[3]),
        (const float*)d_in[14], (const float*)d_in[15], nullptr, E);
    const int n2 = 2 * 24 * 24 * 24 * 128;
    preact_kernel<<<(n2 / 8 + 255) / 256, 256, 0, stream>>>(
        E, (const float*)d_in[17], (const float*)d_in[18], PRE, n2 / 8, 127);
    conv_wmma<128, 128, 3, 1, 1, 24, 24, 24, false, false, false><<<432, 256, 0, stream>>>(
        PRE, (const u16*)(ws + woff[4]), nullptr, nullptr, nullptr, F);
    preact_kernel<<<(n2 / 8 + 255) / 256, 256, 0, stream>>>(
        F, (const float*)d_in[20], (const float*)d_in[21], PRE, n2 / 8, 127);
    conv_wmma<128, 128, 3, 1, 1, 24, 24, 24, false, true, false><<<432, 256, 0, stream>>>(
        PRE, (const u16*)(ws + woff[5]), nullptr, nullptr, E, G);

    // stage 3 ---------------------------------------------------------------
    conv_wmma<128, 128, 2, 2, 0, 12, 12, 12, true, false, false><<<54, 256, 0, stream>>>(
        G, (const u16*)(ws + woff[6]),
        (const float*)d_in[23], (const float*)d_in[24], nullptr, H);
    const int n3 = 2 * 12 * 12 * 12 * 128;
    preact_kernel<<<(n3 / 8 + 255) / 256, 256, 0, stream>>>(
        H, (const float*)d_in[26], (const float*)d_in[27], PRE, n3 / 8, 127);
    conv_wmma<128, 128, 3, 1, 1, 12, 12, 12, false, false, false><<<54, 256, 0, stream>>>(
        PRE, (const u16*)(ws + woff[7]), nullptr, nullptr, nullptr, I);
    preact_kernel<<<(n3 / 8 + 255) / 256, 256, 0, stream>>>(
        I, (const float*)d_in[29], (const float*)d_in[30], PRE, n3 / 8, 127);
    conv_wmma<128, 128, 3, 1, 1, 12, 12, 12, false, true, false><<<54, 256, 0, stream>>>(
        PRE, (const u16*)(ws + woff[8]), nullptr, nullptr, H, J);

    // stage 4 ---------------------------------------------------------------
    conv_wmma<128, 128, 2, 2, 0, 6, 6, 6, true, false, false><<<7, 256, 0, stream>>>(
        J, (const u16*)(ws + woff[9]),
        (const float*)d_in[32], (const float*)d_in[33], nullptr, K);
    const int n4 = 2 * 6 * 6 * 6 * 128;
    preact_kernel<<<(n4 / 8 + 255) / 256, 256, 0, stream>>>(
        K, (const float*)d_in[35], (const float*)d_in[36], PRE, n4 / 8, 127);
    conv_wmma<128, 128, 3, 1, 1, 6, 6, 6, false, false, false><<<7, 256, 0, stream>>>(
        PRE, (const u16*)(ws + woff[10]), nullptr, nullptr, nullptr, Lq);
    preact_kernel<<<(n4 / 8 + 255) / 256, 256, 0, stream>>>(
        Lq, (const float*)d_in[38], (const float*)d_in[39], PRE, n4 / 8, 127);
    conv_wmma<128, 128, 3, 1, 1, 6, 6, 6, false, true, true><<<7, 256, 0, stream>>>(
        PRE, (const u16*)(ws + woff[11]), nullptr, nullptr, K, (float*)d_out);
}
